// L4QQuantizedLinear_79834852098648
// MI455X (gfx1250) — matrix operand order repaired and checked by reading, not compiled, not run on hardware
//
#include <hip/hip_runtime.h>

typedef __attribute__((ext_vector_type(16))) __bf16 v16bf;
typedef __attribute__((ext_vector_type(8)))  float  v8f;

typedef unsigned int       u32;
typedef unsigned short     u16;
typedef unsigned long long u64;

#define IN_F   4096
#define OUT_F  4096
#define RANK   16
#define GROUPQ 128
#define MDIM   8192   // B*S = 4*2048
#define BM     128
#define BN     256
#define BK     64
#define LDT    88     // padded LDS row stride in bf16 elems (176 B: 16B-aligned, conflict-free)

__device__ __forceinline__ u16 f32_to_bf16(float f) {
  u32 u = __builtin_bit_cast(u32, f);
  u32 r = u + 0x7FFFu + ((u >> 16) & 1u);   // round-to-nearest-even
  return (u16)(r >> 16);
}

union FragU {
  uint4 q[2];
  v16bf v;
};

// ---------------------------------------------------------------------------
// Kernel 1: w_q = dequant(quant(w0 + lora_b@lora_a)) -> bf16, into workspace.
// ---------------------------------------------------------------------------
__global__ __launch_bounds__(256) void prep_wq(
    const float* __restrict__ w0, const float* __restrict__ la,
    const float* __restrict__ lb, const float* __restrict__ qs,
    u16* __restrict__ wq) {
  const int o = blockIdx.x;
  __shared__ float lbs[RANK];
  if (threadIdx.x < RANK) lbs[threadIdx.x] = lb[o * RANK + threadIdx.x];
  __syncthreads();

  for (int i = threadIdx.x; i < IN_F; i += 256) {
    float acc = w0[(size_t)o * IN_F + i];
#pragma unroll
    for (int r = 0; r < RANK; ++r)
      acc += lbs[r] * la[r * IN_F + i];          // ALPHA = 1.0
    float s = qs[o * (IN_F / GROUPQ) + (i >> 7)];
    float q = rintf(fminf(fmaxf(acc / s, -8.0f), 7.0f));  // Q_N=-8, Q_P=7
    wq[(size_t)o * IN_F + i] = f32_to_bf16(q * s);
  }
}

// ---------------------------------------------------------------------------
// Kernel 2: out = x @ wq^T + bias   (M=8192, N=4096, K=4096)
// 128x256 block tile, BK=64, 8 wave32s (2x4); wave tile 64x64 = 4x4 WMMA accs.
// A (x): f32 -> bf16 converted in VGPRs while staging to LDS (regular path).
// B (wq): staged with CDNA5 async global->LDS b128 copies (ASYNCcnt).
// ---------------------------------------------------------------------------
__global__ __launch_bounds__(256) void l4q_gemm(
    const float* __restrict__ x, const u16* __restrict__ wq,
    const float* __restrict__ bias, float* __restrict__ out) {
  __shared__ u16 As[BM * LDT];
  __shared__ u16 Bs[BN * LDT];

  const int tid   = threadIdx.x;
  const int lane  = tid & 31;
  const int wave  = tid >> 5;
  const int waveM = wave >> 2;   // 0..1   (64 rows each)
  const int waveN = wave & 3;    // 0..3   (64 cols each)
  const int lo    = lane & 15;
  const int hi    = lane >> 4;

  const int n0 = blockIdx.x * BN;
  const int m0 = blockIdx.y * BM;

  const u32 bsBase = (u32)(size_t)&Bs[0];   // LDS byte offset of Bs (addr[31:0])

  v8f acc[4][4];
  const v8f vzero = {0.f, 0.f, 0.f, 0.f, 0.f, 0.f, 0.f, 0.f};
#pragma unroll
  for (int mt = 0; mt < 4; ++mt)
#pragma unroll
    for (int nt = 0; nt < 4; ++nt) acc[mt][nt] = vzero;

  for (int kt = 0; kt < IN_F; kt += BK) {
    // ---- stage B tile: 256x64 bf16, async global->LDS (2048 b128 slots, 8/thread)
#pragma unroll
    for (int j = 0; j < 8; ++j) {
      int slot = tid + j * 256;
      int row  = slot >> 3;        // / (BK/8)
      int c8   = slot & 7;
      u32 lds  = bsBase + (u32)((row * LDT + c8 * 8) * 2);
      u64 ga   = (u64)(size_t)(wq + (size_t)(n0 + row) * IN_F + kt + c8 * 8);
      asm volatile("global_load_async_to_lds_b128 %0, %1, off"
                   :: "v"(lds), "v"(ga) : "memory");
    }
    // ---- stage A tile: 128x64 f32 -> bf16 (2048 float4 slots, 8/thread)
#pragma unroll
    for (int j = 0; j < 8; ++j) {
      int slot = tid + j * 256;
      int row  = slot >> 4;        // / (BK/4)
      int c4   = slot & 15;
      float4 v = *(const float4*)(x + (size_t)(m0 + row) * IN_F + kt + c4 * 4);
      ushort4 h;
      h.x = f32_to_bf16(v.x); h.y = f32_to_bf16(v.y);
      h.z = f32_to_bf16(v.z); h.w = f32_to_bf16(v.w);
      *(ushort4*)&As[row * LDT + c4 * 4] = h;
    }
    asm volatile("s_wait_asynccnt 0" ::: "memory");
    __syncthreads();

#pragma unroll
    for (int ks = 0; ks < BK; ks += 32) {
      FragU a[4], b[4];
      // A frag (16x32): lane lo = row m; K chunks at hi*8 and hi*8+16
#pragma unroll
      for (int mt = 0; mt < 4; ++mt) {
        const u16* p = &As[(waveM * 64 + mt * 16 + lo) * LDT + ks + hi * 8];
        a[mt].q[0] = *(const uint4*)p;
        a[mt].q[1] = *(const uint4*)(p + 16);
      }
      // B frag (32x16): lane lo = col n; K chunks at hi*8 and hi*8+16
#pragma unroll
      for (int nt = 0; nt < 4; ++nt) {
        const u16* p = &Bs[(waveN * 64 + nt * 16 + lo) * LDT + ks + hi * 8];
        b[nt].q[0] = *(const uint4*)p;
        b[nt].q[1] = *(const uint4*)(p + 16);
      }
#pragma unroll
      for (int mt = 0; mt < 4; ++mt)
#pragma unroll
        for (int nt = 0; nt < 4; ++nt)
          acc[mt][nt] = __builtin_amdgcn_wmma_f32_16x16x32_bf16(
              false, a[mt].v, false, b[nt].v, (short)0, acc[mt][nt],
              false, false);
    }
    __syncthreads();
  }

  // ---- epilogue: C layout -> VGPR r holds row mt*16 + r + hi*8, col = lo
#pragma unroll
  for (int nt = 0; nt < 4; ++nt) {
    int col  = n0 + waveN * 64 + nt * 16 + lo;
    float bv = bias[col];
#pragma unroll
    for (int mt = 0; mt < 4; ++mt) {
      int rowb = m0 + waveM * 64 + mt * 16 + hi * 8;
#pragma unroll
      for (int r = 0; r < 8; ++r)
        out[(size_t)(rowb + r) * OUT_F + col] = acc[mt][nt][r] + bv;
    }
  }
}

// ---------------------------------------------------------------------------
extern "C" void kernel_launch(void* const* d_in, const int* in_sizes, int n_in,
                              void* d_out, int out_size, void* d_ws, size_t ws_size,
                              hipStream_t stream) {
  (void)in_sizes; (void)n_in; (void)out_size; (void)ws_size;
  const float* x    = (const float*)d_in[0];  // (4,2048,4096)
  const float* w0   = (const float*)d_in[1];  // (4096,4096)
  const float* la   = (const float*)d_in[2];  // (16,4096)
  const float* lb   = (const float*)d_in[3];  // (4096,16)
  const float* qs   = (const float*)d_in[4];  // (131072,)
  const float* bias = (const float*)d_in[5];  // (4096,)
  float* out = (float*)d_out;
  u16*   wq  = (u16*)d_ws;                    // 4096*4096*2 B = 33.5 MB

  prep_wq<<<OUT_F, 256, 0, stream>>>(w0, la, lb, qs, wq);

  dim3 grid(OUT_F / BN, MDIM / BM);           // (16, 64)
  l4q_gemm<<<grid, 256, 0, stream>>>(x, wq, bias, out);
}
